// WaveNet_43396349559249
// MI455X (gfx1250) — matrix operand order repaired and checked by reading, not compiled, not run on hardware
//
#include <hip/hip_runtime.h>
#include <string.h>

// ---------------------------------------------------------------------------
// Types for CDNA5 WMMA
// ---------------------------------------------------------------------------
typedef __attribute__((ext_vector_type(16))) __bf16         v16bf;
typedef __attribute__((ext_vector_type(8)))  float          v8f;
typedef __attribute__((ext_vector_type(8)))  unsigned short v8us;

#define NB       4       // batch
#define TWAV     8192
#define SIG_TP   8256    // 8192 + tile pad
#define COND_TP  8256
#define SKP_T    6146
#define SKP_TP   6208    // 97 tiles * 64
#define TTILE    64
#define TOTALDIL 2046

static __device__ __forceinline__ unsigned short f2bf_u(float f) {
    union { float f; unsigned u; } c; c.f = f;
    unsigned u = c.u;
    return (unsigned short)((u + 0x7FFFu + ((u >> 16) & 1u)) >> 16);
}
static __device__ __forceinline__ float bf2f_u(unsigned short s) {
    union { unsigned u; float f; } c; c.u = ((unsigned)s) << 16;
    return c.f;
}
static __device__ __forceinline__ v8f zero8() {
    v8f z = {0.f,0.f,0.f,0.f,0.f,0.f,0.f,0.f};
    return z;
}
#define WMMA_BF16(A,Bv,C) \
    __builtin_amdgcn_wmma_f32_16x16x32_bf16(false, (A), false, (Bv), (short)0, (C), false, false)

// Async copy of one 16-byte chunk global->LDS (gfx1250, ASYNCcnt-tracked).
// LDS byte address = low 32 bits of the generic shared pointer.
static __device__ __forceinline__ void async_g2l_b128(void* lds_ptr, const void* gptr) {
    unsigned lds_off = (unsigned)(size_t)lds_ptr;
    asm volatile("global_load_async_to_lds_b128 %0, %1, off"
                 :: "v"(lds_off), "v"(gptr) : "memory");
}
static __device__ __forceinline__ void wait_async0() {
    asm volatile("s_wait_asynccnt 0x0" ::: "memory");
}

// ---------------------------------------------------------------------------
// Weight packing into WMMA A-fragment layout (contiguous k order, matching the B-operand loads):
// lane l, element e:  M = l%16,  K = (l/16)*16 + e
// dst block (mtile*nKt + kt): 32 lanes * 16 bf16 contiguous.
// ---------------------------------------------------------------------------
__global__ void pack_kernel(const float* __restrict__ src, unsigned short* __restrict__ dst,
                            int nKt, long rowStride, long elemStride,
                            long srcLayerStride, long dstLayerStride) {
    int l   = threadIdx.x;
    int blk = blockIdx.x;
    src += (size_t)blockIdx.y * srcLayerStride;
    dst += (size_t)blockIdx.y * dstLayerStride;
    int mtile = blk / nKt, kt = blk % nKt;
    int M = mtile * 16 + (l & 15);
    unsigned short* d = dst + (size_t)blk * 512 + l * 16;
#pragma unroll
    for (int e = 0; e < 16; ++e) {
        int K = kt * 32 + ((l >> 4) << 4) + e;
        d[e] = f2bf_u(src[(size_t)M * rowStride + (size_t)K * elemStride]);
    }
}

// Wt_up (4,128,128,8) -> w2[s][k][i][o] for coalesced upsample reads
__global__ void pack_up_kernel(const float* __restrict__ wt, float* __restrict__ w2) {
    int o = threadIdx.x, s = blockIdx.x, k = blockIdx.y, i = blockIdx.z;
    w2[(((size_t)s * 8 + k) * 128 + i) * 128 + o] =
        wt[(((size_t)s * 128 + i) * 128 + o) * 8 + k];
}

// ---------------------------------------------------------------------------
// lc gather (jitter) + k=3 valid conv : out (B,128,34) f32 channel-major
// ---------------------------------------------------------------------------
__global__ __launch_bounds__(128) void lc_kernel(
        const float* __restrict__ lc_sparse, const int* __restrict__ jit,
        const float* __restrict__ Wlc, const float* __restrict__ blc,
        float* __restrict__ out) {
    __shared__ float xs[3][64];
    int o = threadIdx.x, t = blockIdx.x, b = blockIdx.y;
    for (int idx = o; idx < 192; idx += 128) {
        int k = idx / 64, i = idx % 64;
        int jj = jit[b * 36 + t + k];
        xs[k][i] = lc_sparse[((size_t)b * 64 + i) * 36 + jj];
    }
    __syncthreads();
    float acc = blc[o];
    for (int i = 0; i < 64; ++i)
#pragma unroll
        for (int k = 0; k < 3; ++k)
            acc += Wlc[((size_t)o * 64 + i) * 3 + k] * xs[k][i];
    out[((size_t)b * 128 + o) * 34 + t] = acc;
}

// ---------------------------------------------------------------------------
// conv_transpose1d stride4 filt8 (reference padding) : y[t,o], taps k0 & k0+4
// ---------------------------------------------------------------------------
__global__ __launch_bounds__(128) void upsample_kernel(
        const float* __restrict__ x, float* __restrict__ y,
        const float* __restrict__ w2 /*[8][128][128]*/, const float* __restrict__ bup,
        int Tin, int Tout) {
    __shared__ float x0[128], x1[128];
    int o = threadIdx.x, b = blockIdx.y;
    const float* xb = x + (size_t)b * 128 * Tin;
    int tend = blockIdx.x * 32 + 32; if (tend > Tout) tend = Tout;
    for (int t = blockIdx.x * 32; t < tend; ++t) {
        int k0 = ((3 - t) % 4 + 4) % 4;
        int j0 = (t + k0 - 3) >> 2;          // always >= 0
        __syncthreads();
        x0[o] = (j0     < Tin) ? xb[(size_t)o * Tin + j0]     : 0.f;
        x1[o] = (j0 + 1 < Tin) ? xb[(size_t)o * Tin + j0 + 1] : 0.f;
        __syncthreads();
        float acc = bup[o];
        const float* wA = w2 + (size_t)(7 - k0) * 16384 + o;       // tap k0
        const float* wB = w2 + (size_t)(3 - k0) * 16384 + o;       // tap k0+4
        for (int i = 0; i < 128; ++i)
            acc += wA[(size_t)i * 128] * x0[i] + wB[(size_t)i * 128] * x1[i];
        y[((size_t)b * 128 + o) * Tout + t] = acc;
    }
}

// ---------------------------------------------------------------------------
// cond build: lc part transposed (trim 86) + gc broadcast. cond bf16 [B][T][192]
// ---------------------------------------------------------------------------
__global__ void cond_lc_kernel(const float* __restrict__ lc4, unsigned short* __restrict__ cond) {
    __shared__ float tile[32][33];
    int b = blockIdx.z, c0 = blockIdx.y * 32, t0 = blockIdx.x * 32;
    for (int yy = threadIdx.y; yy < 32; yy += 8) {
        int c = c0 + yy, t = t0 + threadIdx.x;
        tile[yy][threadIdx.x] = (t < TWAV - 1) ? lc4[((size_t)b * 128 + c) * 8364 + 86 + t] : 0.f;
    }
    __syncthreads();
    for (int yy = threadIdx.y; yy < 32; yy += 8) {
        int t = t0 + yy, c = c0 + threadIdx.x;
        if (t < TWAV - 1)
            cond[((size_t)b * COND_TP + t) * 192 + c] = f2bf_u(tile[threadIdx.x][yy]);
    }
}
__global__ void cond_gc_kernel(const float* __restrict__ Wspk, const float* __restrict__ bspk,
                               const int* __restrict__ spk, unsigned short* __restrict__ cond) {
    int g = threadIdx.x, b = blockIdx.y;
    unsigned short u = f2bf_u(Wspk[(size_t)g * 40 + spk[b]] + bspk[g]);
    int t0 = blockIdx.x * 128;
    for (int t = t0; t < t0 + 128 && t < TWAV - 1; ++t)
        cond[((size_t)b * COND_TP + t) * 192 + 128 + g] = u;
}

// wav (B,256,8192) f32 channel-major -> wavT [B][8192][256] bf16
__global__ void wav_tr_kernel(const float* __restrict__ wav, unsigned short* __restrict__ wavT) {
    __shared__ float tile[32][33];
    int b = blockIdx.z, c0 = blockIdx.y * 32, t0 = blockIdx.x * 32;
    for (int yy = threadIdx.y; yy < 32; yy += 8)
        tile[yy][threadIdx.x] = wav[((size_t)b * 256 + c0 + yy) * TWAV + t0 + threadIdx.x];
    __syncthreads();
    for (int yy = threadIdx.y; yy < 32; yy += 8)
        wavT[((size_t)b * TWAV + t0 + yy) * 256 + c0 + threadIdx.x] =
            f2bf_u(tile[threadIdx.x][yy]);
}

__global__ void relu_bf16_kernel(const float* __restrict__ in, unsigned short* __restrict__ out,
                                 size_t n) {
    size_t i = (size_t)blockIdx.x * 256 + threadIdx.x;
    if (i < n) out[i] = f2bf_u(fmaxf(in[i], 0.f));
}

// ---------------------------------------------------------------------------
// Generic time-major GEMM: D[b][t][m] = act( A(MxK) @ actv[b][t][k] + bias )
// block = 8 waves; each wave owns mtPerWave 16-row M tiles, 4 N tiles (64 t).
// mode 0: bf16 TM out; 1: relu->bf16 TM out; 2: f32 channel-major out
// ---------------------------------------------------------------------------
__global__ __launch_bounds__(256) void gemm_tm_kernel(
        const unsigned short* __restrict__ Apk, int nKt, int mtPerWave,
        const unsigned short* __restrict__ act, int Cin, int TpadIn,
        const float* __restrict__ bias,
        void* __restrict__ outPtr, int Cout, int TpadOut, int Tvalid, int mode) {
    const int wave = threadIdx.x >> 5, lane = threadIdx.x & 31;
    const int nhi = lane >> 4, nlo = lane & 15;
    const int b = blockIdx.y;
    const int t0 = blockIdx.x * TTILE;

    for (int j = 0; j < mtPerWave; ++j) {
        int mtile = wave * mtPerWave + j;
        v8f acc[4];
#pragma unroll
        for (int nt = 0; nt < 4; ++nt) acc[nt] = zero8();
        const unsigned short* Ab = Apk + (size_t)mtile * nKt * 512;
        for (int kt = 0; kt < nKt; ++kt) {
            v16bf a = *(const v16bf*)(Ab + kt * 512 + lane * 16);
#pragma unroll
            for (int nt = 0; nt < 4; ++nt) {
                const unsigned short* bp =
                    act + ((size_t)b * TpadIn + t0 + nt * 16 + nlo) * Cin + kt * 32 + nhi * 16;
                v16bf bf = *(const v16bf*)bp;
                acc[nt] = WMMA_BF16(a, bf, acc[nt]);
            }
        }
        int mbase = mtile * 16 + nhi * 8;
        float bz[8];
#pragma unroll
        for (int v = 0; v < 8; ++v) bz[v] = bias[mbase + v];
#pragma unroll
        for (int nt = 0; nt < 4; ++nt) {
            int t = t0 + nt * 16 + nlo;
            if (t < Tvalid) {
                if (mode == 2) {
                    float* o = (float*)outPtr + ((size_t)b * Cout + mbase) * (size_t)Tvalid + t;
#pragma unroll
                    for (int v = 0; v < 8; ++v) o[(size_t)v * Tvalid] = acc[nt][v] + bz[v];
                } else {
                    unsigned short* o =
                        (unsigned short*)outPtr + ((size_t)b * TpadOut + t) * Cout + mbase;
                    v8us pk;
#pragma unroll
                    for (int v = 0; v < 8; ++v) {
                        float x = acc[nt][v] + bz[v];
                        if (mode == 1) x = fmaxf(x, 0.f);
                        pk[v] = f2bf_u(x);
                    }
                    *(v8us*)o = pk;   // one 16B store, 8 consecutive channels
                }
            }
        }
    }
}

// ---------------------------------------------------------------------------
// Fused WaveNet layer. End-aligned absolute time t:
//   filt/gate = Wsig0@sig[t-d] + Wsig1@sig[t] + Wp@cond[t-1] + bias
//   z = tanh(filt)*sigmoid(gate)  (staged to LDS, time-major bf16)
//   skpSum[t-2046] += Wskp@z      (t >= 2046)
//   sigOut[t]      = Wres@z + sigIn[t]
// B operands are staged ONCE per block into LDS via async global->LDS DMA
// (ASYNCcnt), removing the 8x per-wave redundant global fetch.
// ---------------------------------------------------------------------------
__global__ __launch_bounds__(256) void wavenet_layer_kernel(
        const unsigned short* __restrict__ sigIn, unsigned short* __restrict__ sigOut,
        const unsigned short* __restrict__ cond, float* __restrict__ skpSum,
        const unsigned short* __restrict__ Asig0, const unsigned short* __restrict__ Asig1,
        const unsigned short* __restrict__ Agate0, const unsigned short* __restrict__ Agate1,
        const unsigned short* __restrict__ Apsig, const unsigned short* __restrict__ Apgate,
        const unsigned short* __restrict__ Ares, const unsigned short* __restrict__ Askp,
        const float* __restrict__ bsig, const float* __restrict__ bgate,
        int d, int tStart) {
    __shared__ __align__(32) unsigned short s0lds[TTILE * 128];  // sig[t-d] tile, 16KB
    __shared__ __align__(32) unsigned short s1lds[TTILE * 128];  // sig[t]   tile, 16KB
    __shared__ __align__(32) unsigned short clds[TTILE * 192];   // cond[t-1] tile, 24KB
    __shared__ __align__(32) unsigned short zlds[TTILE * 128];   // z tile,  16KB
    const int tid  = threadIdx.x;
    const int wave = tid >> 5, lane = tid & 31;
    const int nhi = lane >> 4, nlo = lane & 15;
    const int b = blockIdx.y;
    const int t0 = tStart + blockIdx.x * TTILE;

    // ---- async-stage B tiles (linear copies; chunk counts are exact multiples
    //      of 256 threads -> no EXEC divergence around async ops) ----
    {
        const unsigned short* g0 = sigIn + ((size_t)b * SIG_TP + (t0 - d)) * 128;
        const unsigned short* g1 = sigIn + ((size_t)b * SIG_TP + t0) * 128;
        const unsigned short* gc = cond  + ((size_t)b * COND_TP + (t0 - 1)) * 192;
#pragma unroll
        for (int it = 0; it < 4; ++it) {       // 1024 chunks of 16B = 16KB
            int idx = it * 256 + tid;
            async_g2l_b128(&s0lds[idx * 8], g0 + idx * 8);
        }
#pragma unroll
        for (int it = 0; it < 4; ++it) {
            int idx = it * 256 + tid;
            async_g2l_b128(&s1lds[idx * 8], g1 + idx * 8);
        }
#pragma unroll
        for (int it = 0; it < 6; ++it) {       // 1536 chunks = 24KB
            int idx = it * 256 + tid;
            async_g2l_b128(&clds[idx * 8], gc + idx * 8);
        }
        wait_async0();
        __syncthreads();
    }

    // ---- phase 1: filt + gate, K = 448 in 14 k-tiles, shared B operands ----
    const int mtile = wave;
    v8f accF[4], accG[4];
#pragma unroll
    for (int nt = 0; nt < 4; ++nt) { accF[nt] = zero8(); accG[nt] = zero8(); }
#pragma unroll
    for (int kt = 0; kt < 14; ++kt) {
        const unsigned short *Af, *Ag, *bb;
        int C;
        if (kt < 4) {
            Af = Asig0  + (mtile * 4 + kt) * 512;
            Ag = Agate0 + (mtile * 4 + kt) * 512;
            bb = &s0lds[kt * 32];                          C = 128;
        } else if (kt < 8) {
            int kk = kt - 4;
            Af = Asig1  + (mtile * 4 + kk) * 512;
            Ag = Agate1 + (mtile * 4 + kk) * 512;
            bb = &s1lds[kk * 32];                          C = 128;
        } else {
            int kk = kt - 8;
            Af = Apsig  + (mtile * 6 + kk) * 512;
            Ag = Apgate + (mtile * 6 + kk) * 512;
            bb = &clds[kk * 32];                           C = 192;
        }
        v16bf af = *(const v16bf*)(Af + lane * 16);
        v16bf ag = *(const v16bf*)(Ag + lane * 16);
#pragma unroll
        for (int nt = 0; nt < 4; ++nt) {
            v16bf bf = *(const v16bf*)(bb + (size_t)(nt * 16 + nlo) * C + nhi * 16);
            accF[nt] = WMMA_BF16(af, bf, accF[nt]);
            accG[nt] = WMMA_BF16(ag, bf, accG[nt]);
        }
    }

    // ---- gated activation -> z tile in LDS (time-major bf16 [64][128]) ----
    const int mbase = mtile * 16 + nhi * 8;
    float bsv[8], bgv[8];
#pragma unroll
    for (int v = 0; v < 8; ++v) { bsv[v] = bsig[mbase + v]; bgv[v] = bgate[mbase + v]; }
#pragma unroll
    for (int nt = 0; nt < 4; ++nt) {
        v8us zr;
#pragma unroll
        for (int v = 0; v < 8; ++v) {
            float f = accF[nt][v] + bsv[v];
            float g = accG[nt][v] + bgv[v];
            float z = tanhf(f) * (1.0f / (1.0f + __expf(-g)));
            zr[v] = f2bf_u(z);
        }
        *(v8us*)&zlds[(nt * 16 + nlo) * 128 + mbase] = zr;
    }
    __syncthreads();

    // ---- phase 2: skip (16 M-tiles) + residual (8 M-tiles), K = 128 ----
#pragma unroll
    for (int j = 0; j < 3; ++j) {
        int vm = wave * 3 + j;
        const unsigned short* Ap =
            (vm < 16) ? (Askp + (size_t)vm * 4 * 512) : (Ares + (size_t)(vm - 16) * 4 * 512);
        v8f acc[4];
#pragma unroll
        for (int nt = 0; nt < 4; ++nt) acc[nt] = zero8();
#pragma unroll
        for (int kt = 0; kt < 4; ++kt) {
            v16bf a = *(const v16bf*)(Ap + kt * 512 + lane * 16);
#pragma unroll
            for (int nt = 0; nt < 4; ++nt) {
                v16bf bf = *(const v16bf*)&zlds[(nt * 16 + nlo) * 128 + kt * 32 + nhi * 16];
                acc[nt] = WMMA_BF16(a, bf, acc[nt]);
            }
        }
        if (vm < 16) {
            int cb = vm * 16 + nhi * 8;
#pragma unroll
            for (int nt = 0; nt < 4; ++nt) {
                int t = t0 + nt * 16 + nlo;
                if (t >= TOTALDIL && t < TWAV) {
                    float* p = skpSum + ((size_t)b * SKP_TP + (t - TOTALDIL)) * 256 + cb;
                    float4 lo = *(float4*)p, hi = *(float4*)(p + 4);
                    lo.x += acc[nt][0]; lo.y += acc[nt][1];
                    lo.z += acc[nt][2]; lo.w += acc[nt][3];
                    hi.x += acc[nt][4]; hi.y += acc[nt][5];
                    hi.z += acc[nt][6]; hi.w += acc[nt][7];
                    *(float4*)p = lo; *(float4*)(p + 4) = hi;
                }
            }
        } else {
            int cb = (vm - 16) * 16 + nhi * 8;
#pragma unroll
            for (int nt = 0; nt < 4; ++nt) {
                int t = t0 + nt * 16 + nlo;
                if (t < TWAV) {
                    size_t idx = ((size_t)b * SIG_TP + t) * 128 + cb;
                    v8us old = *(const v8us*)&sigIn[idx];
                    v8us nw;
#pragma unroll
                    for (int v = 0; v < 8; ++v)
                        nw[v] = f2bf_u(acc[nt][v] + bf2f_u(old[v]));
                    *(v8us*)&sigOut[idx] = nw;
                }
            }
        }
    }
}

// ---------------------------------------------------------------------------
// Host side
// ---------------------------------------------------------------------------
extern "C" void kernel_launch(void* const* d_in, const int* in_sizes, int n_in,
                              void* d_out, int out_size, void* d_ws, size_t ws_size,
                              hipStream_t stream) {
    const float* wav       = (const float*)d_in[0];
    const float* lc_sparse = (const float*)d_in[1];
    const int*   spk       = (const int*)  d_in[2];
    const int*   jit       = (const int*)  d_in[3];
    const float* W_lc      = (const float*)d_in[4];
    const float* b_lc      = (const float*)d_in[5];
    const float* Wt_up     = (const float*)d_in[6];
    const float* b_up      = (const float*)d_in[7];
    const float* W_spk     = (const float*)d_in[8];
    const float* b_spk     = (const float*)d_in[9];
    const float* W_base    = (const float*)d_in[10];
    const float* b_base    = (const float*)d_in[11];
    const float* W_sig     = (const float*)d_in[12];
    const float* b_sig     = (const float*)d_in[13];
    const float* W_gate    = (const float*)d_in[14];
    const float* b_gate    = (const float*)d_in[15];
    const float* W_psig    = (const float*)d_in[16];
    const float* W_pgate   = (const float*)d_in[17];
    const float* W_res     = (const float*)d_in[18];
    const float* W_skp     = (const float*)d_in[19];
    const float* W_post1   = (const float*)d_in[20];
    const float* b_post1   = (const float*)d_in[21];
    const float* W_post2   = (const float*)d_in[22];
    const float* b_post2   = (const float*)d_in[23];

    static const int DILS[20] = {1,2,4,8,16,32,64,128,256,512,
                                 1,2,4,8,16,32,64,128,256,512};
    int CUM[20]; { int s = 0; for (int i = 0; i < 20; ++i) { s += DILS[i]; CUM[i] = s; } }

    char* ws = (char*)d_ws;
    size_t off = 0;
    auto carve = [&](size_t bytes) -> void* {
        void* p = ws + off;
        off = (off + bytes + 255) & ~(size_t)255;
        return p;
    };
    unsigned short* sigA   = (unsigned short*)carve((size_t)NB * SIG_TP * 128 * 2);
    unsigned short* sigB   = (unsigned short*)carve((size_t)NB * SIG_TP * 128 * 2);
    unsigned short* cond   = (unsigned short*)carve((size_t)NB * COND_TP * 192 * 2);
    unsigned short* wavT   = (unsigned short*)carve((size_t)NB * TWAV * 256 * 2);
    float*          skpSum = (float*)        carve((size_t)NB * SKP_TP * 256 * 4);
    unsigned short* skpb   = (unsigned short*)carve((size_t)NB * SKP_TP * 256 * 2);
    unsigned short* post1b = (unsigned short*)carve((size_t)NB * SKP_TP * 256 * 2);
    float*          lc0    = (float*)carve((size_t)NB * 128 * 34 * 4);
    float*          upA    = (float*)carve((size_t)NB * 128 * 8364 * 4);
    float*          upB    = (float*)carve((size_t)NB * 128 * 8364 * 4);
    float*          w2up   = (float*)carve((size_t)4 * 8 * 128 * 128 * 4);
    unsigned short* pk_sig0  = (unsigned short*)carve((size_t)20 * 16384 * 2);
    unsigned short* pk_sig1  = (unsigned short*)carve((size_t)20 * 16384 * 2);
    unsigned short* pk_gate0 = (unsigned short*)carve((size_t)20 * 16384 * 2);
    unsigned short* pk_gate1 = (unsigned short*)carve((size_t)20 * 16384 * 2);
    unsigned short* pk_psig  = (unsigned short*)carve((size_t)20 * 24576 * 2);
    unsigned short* pk_pgate = (unsigned short*)carve((size_t)20 * 24576 * 2);
    unsigned short* pk_res   = (unsigned short*)carve((size_t)20 * 16384 * 2);
    unsigned short* pk_skp   = (unsigned short*)carve((size_t)20 * 32768 * 2);
    unsigned short* pk_base  = (unsigned short*)carve((size_t)32768 * 2);
    unsigned short* pk_post1 = (unsigned short*)carve((size_t)65536 * 2);
    unsigned short* pk_post2 = (unsigned short*)carve((size_t)65536 * 2);

    // ---- weight packing ----
    pack_kernel<<<dim3(32, 20), 32, 0, stream>>>(W_sig,     pk_sig0,  4, 256, 2, 32768, 16384);
    pack_kernel<<<dim3(32, 20), 32, 0, stream>>>(W_sig + 1, pk_sig1,  4, 256, 2, 32768, 16384);
    pack_kernel<<<dim3(32, 20), 32, 0, stream>>>(W_gate,    pk_gate0, 4, 256, 2, 32768, 16384);
    pack_kernel<<<dim3(32, 20), 32, 0, stream>>>(W_gate + 1,pk_gate1, 4, 256, 2, 32768, 16384);
    pack_kernel<<<dim3(48, 20), 32, 0, stream>>>(W_psig,    pk_psig,  6, 192, 1, 24576, 24576);
    pack_kernel<<<dim3(48, 20), 32, 0, stream>>>(W_pgate,   pk_pgate, 6, 192, 1, 24576, 24576);
    pack_kernel<<<dim3(32, 20), 32, 0, stream>>>(W_res,     pk_res,   4, 128, 1, 16384, 16384);
    pack_kernel<<<dim3(64, 20), 32, 0, stream>>>(W_skp,     pk_skp,   4, 128, 1, 32768, 32768);
    pack_kernel<<<dim3(64, 1),  32, 0, stream>>>(W_base,    pk_base,  8, 256, 1, 0, 0);
    pack_kernel<<<dim3(128, 1), 32, 0, stream>>>(W_post1,   pk_post1, 8, 256, 1, 0, 0);
    pack_kernel<<<dim3(128, 1), 32, 0, stream>>>(W_post2,   pk_post2, 8, 256, 1, 0, 0);
    pack_up_kernel<<<dim3(4, 8, 128), 128, 0, stream>>>(Wt_up, w2up);

    hipMemsetAsync(skpSum, 0, (size_t)NB * SKP_TP * 256 * 4, stream);

    // ---- conditioning path ----
    lc_kernel<<<dim3(34, NB), 128, 0, stream>>>(lc_sparse, jit, W_lc, b_lc, lc0);
    const int TOUT[4] = {132, 524, 2092, 8364};
    const float* upin[4] = {lc0, upA, upB, upA};
    float* upout[4]      = {upA, upB, upA, upB};
    int Tin = 34;
    for (int s = 0; s < 4; ++s) {
        upsample_kernel<<<dim3((TOUT[s] + 31) / 32, NB), 128, 0, stream>>>(
            upin[s], upout[s], w2up + (size_t)s * 8 * 128 * 128, b_up + s * 128, Tin, TOUT[s]);
        Tin = TOUT[s];
    }
    cond_lc_kernel<<<dim3(256, 4, NB), dim3(32, 8), 0, stream>>>(upB, cond);
    cond_gc_kernel<<<dim3(64, NB), 64, 0, stream>>>(W_spk, b_spk, spk, cond);

    // ---- base GEMM: wav one-hot -> sig ----
    wav_tr_kernel<<<dim3(256, 8, NB), dim3(32, 8), 0, stream>>>(wav, wavT);
    gemm_tm_kernel<<<dim3(TWAV / TTILE, NB), 256, 0, stream>>>(
        pk_base, 8, 1, wavT, 256, TWAV, b_base, sigA, 128, SIG_TP, TWAV, 0);

    // ---- 20 fused WaveNet layers (ping-pong sig buffers) ----
    for (int i = 0; i < 20; ++i) {
        const unsigned short* sin = (i & 1) ? sigB : sigA;
        unsigned short*       sout = (i & 1) ? sigA : sigB;
        int tiles = (TWAV - CUM[i] + TTILE - 1) / TTILE;
        wavenet_layer_kernel<<<dim3(tiles, NB), 256, 0, stream>>>(
            sin, sout, cond, skpSum,
            pk_sig0 + (size_t)i * 16384, pk_sig1 + (size_t)i * 16384,
            pk_gate0 + (size_t)i * 16384, pk_gate1 + (size_t)i * 16384,
            pk_psig + (size_t)i * 24576, pk_pgate + (size_t)i * 24576,
            pk_res + (size_t)i * 16384, pk_skp + (size_t)i * 32768,
            b_sig + i * 128, b_gate + i * 128, DILS[i], CUM[i]);
    }

    // ---- post network ----
    size_t nskp = (size_t)NB * SKP_TP * 256;
    relu_bf16_kernel<<<dim3((unsigned)((nskp + 255) / 256)), 256, 0, stream>>>(skpSum, skpb, nskp);
    gemm_tm_kernel<<<dim3(SKP_TP / TTILE, NB), 256, 0, stream>>>(
        pk_post1, 8, 2, skpb, 256, SKP_TP, b_post1, post1b, 256, SKP_TP, SKP_T, 1);
    gemm_tm_kernel<<<dim3(SKP_TP / TTILE, NB), 256, 0, stream>>>(
        pk_post2, 8, 2, post1b, 256, SKP_TP, b_post2, d_out, 256, 0, SKP_T, 2);
}
